// DINOv3ViTAttention_10754598109638
// MI455X (gfx1250) — matrix operand order repaired and checked by reading, not compiled, not run on hardware
//
#include <hip/hip_runtime.h>
#include <hip/hip_bf16.h>

typedef __bf16 bf16_t;
typedef __attribute__((ext_vector_type(16))) __bf16 v16bf;
typedef __attribute__((ext_vector_type(8)))  __bf16 v8bf;
typedef __attribute__((ext_vector_type(4)))  __bf16 v4bf;
typedef __attribute__((ext_vector_type(8)))  float  v8f;

#define NB   16
#define NSEQ 1029
#define NH   16
#define HDIM 64
#define DMODEL 1024
#define PREFIX 5
#define NPAD 1056            // 33 * 32
#define MTOT (NB * NSEQ)     // 16464 = 343 * 48 = 1029 * 16

static __device__ __forceinline__ v8f wmma_bf16(v16bf a, v16bf b, v8f c) {
    // D = A(16x32 bf16) * B(32x16 bf16) + C(16x16 f32)
    return __builtin_amdgcn_wmma_f32_16x16x32_bf16(false, a, false, b,
                                                   (short)0, c, false, false);
}

union V16U { v16bf v; v8bf h[2]; };

static __device__ __forceinline__ v8bf ld8(const bf16_t* p) {
    return *reinterpret_cast<const v8bf*>(p);
}
static __device__ __forceinline__ v16bf mk16(v8bf lo, v8bf hi) {
    V16U u; u.h[0] = lo; u.h[1] = hi; return u.v;
}
// A-matrix (16x32, row = lane%16, contraction contiguous in memory):
// lanes 0-15 load K = [kb..kb+15] (code's order; B matches); lanes 16-31 K = [kb+16..kb+31].
static __device__ __forceinline__ v16bf ldA(const bf16_t* row, int kb, int hf) {
    return mk16(ld8(row + kb + hf * 16), ld8(row + kb + hf * 16 + 8));
}
// B-matrix (32x16): per lane 16 contiguous contraction elements (caller adds hf*16).
static __device__ __forceinline__ v16bf ldB(const bf16_t* p) {
    return mk16(ld8(p), ld8(p + 8));
}

// ---------------------------------------------------------------- convert
__global__ __launch_bounds__(256) void cvt_bf16_kernel(const float* __restrict__ src,
                                                       bf16_t* __restrict__ dst, int n4) {
    int i = blockIdx.x * blockDim.x + threadIdx.x;
    if (i < n4) {
        float4 f = reinterpret_cast<const float4*>(src)[i];
        v4bf o = { (bf16_t)f.x, (bf16_t)f.y, (bf16_t)f.z, (bf16_t)f.w };
        reinterpret_cast<v4bf*>(dst)[i] = o;
    }
}

// ---------------------------------------------------------------- QKV proj + RoPE
// grid (343, 6), block 256 (8 waves). wave job = which(q/k/v) x head.
// Each wave: 48 rows (3 M-tiles) x 64 cols (one full head) of one of Q/K/V.
// B tiles reused across the 3 M-tiles in registers; A shared across waves via L1.
__global__ __launch_bounds__(256) void qkv_proj_kernel(
    const bf16_t* __restrict__ Xbf, const bf16_t* __restrict__ Wbf,
    const float* __restrict__ bq, const float* __restrict__ bk, const float* __restrict__ bv,
    const float* __restrict__ sinp, const float* __restrict__ cosp,
    bf16_t* __restrict__ Q, bf16_t* __restrict__ K, bf16_t* __restrict__ VT)
{
    const int lane = threadIdx.x & 31;
    const int wv   = threadIdx.x >> 5;
    const int l16  = lane & 15;
    const int hf   = lane >> 4;
    const int job  = blockIdx.y * 8 + wv;   // 0..47
    const int which = job >> 4;             // 0=q 1=k 2=v
    const int head  = job & 15;
    const int m0    = blockIdx.x * 48;

    const bf16_t* W = Wbf + (size_t)which * (DMODEL * DMODEL);

    v8f acc[3][4] = {};
    for (int k0 = 0; k0 < DMODEL; k0 += 32) {
        v16bf a[3];
#pragma unroll
        for (int mt = 0; mt < 3; ++mt)
            a[mt] = ldA(Xbf + (size_t)(m0 + mt * 16 + l16) * DMODEL, k0, hf);
#pragma unroll
        for (int t = 0; t < 4; ++t) {
            const bf16_t* wrow = W + (size_t)(head * 64 + t * 16 + l16) * DMODEL + k0 + hf * 16;
            v16bf b = ldB(wrow);
#pragma unroll
            for (int mt = 0; mt < 3; ++mt)
                acc[mt][t] = wmma_bf16(a[mt], b, acc[mt][t]);
        }
    }

    const float* bias = (which == 0) ? bq : (which == 1) ? bk : bv;
#pragma unroll
    for (int t = 0; t < 4; ++t) {
        float bvv = bias[head * 64 + t * 16 + l16];
#pragma unroll
        for (int mt = 0; mt < 3; ++mt)
#pragma unroll
            for (int r = 0; r < 8; ++r) acc[mt][t][r] += bvv;
    }

    // RoPE on q,k. C-layout: reg r of tile mt holds row m0+mt*16+r(+8 for hi lanes),
    // col = head*64 + t*16 + l16. Pairs (d, d+32) live in acc[.][t] / acc[.][t+2].
    if (which < 2) {
#pragma unroll
        for (int mt = 0; mt < 3; ++mt)
#pragma unroll
            for (int r = 0; r < 8; ++r) {
                int mrow = m0 + mt * 16 + r + hf * 8;
                int bb = mrow / NSEQ;
                int nn = mrow - bb * NSEQ;
                int p  = nn - PREFIX;
                if (p >= 0) {
#pragma unroll
                    for (int t = 0; t < 2; ++t) {
                        int d = t * 16 + l16;
                        float c1 = cosp[p * HDIM + d],      s1 = sinp[p * HDIM + d];
                        float c2 = cosp[p * HDIM + d + 32], s2 = sinp[p * HDIM + d + 32];
                        float x1 = acc[mt][t][r], x2 = acc[mt][t + 2][r];
                        acc[mt][t][r]     = x1 * c1 - x2 * s1;
                        acc[mt][t + 2][r] = x2 * c2 + x1 * s2;
                    }
                }
            }
    }

    if (which == 2) { // V stored transposed: [b,h, d, n]
#pragma unroll
        for (int mt = 0; mt < 3; ++mt)
#pragma unroll
            for (int r = 0; r < 8; ++r) {
                int mrow = m0 + mt * 16 + r + hf * 8;
                int bb = mrow / NSEQ;
                int nn = mrow - bb * NSEQ;
                bf16_t* base = VT + (size_t)(bb * NH + head) * HDIM * NPAD;
#pragma unroll
                for (int t = 0; t < 4; ++t)
                    base[(size_t)(t * 16 + l16) * NPAD + nn] = (bf16_t)acc[mt][t][r];
            }
    } else {          // Q/K stored [b,h, n, d]
        bf16_t* dst = (which == 0) ? Q : K;
#pragma unroll
        for (int mt = 0; mt < 3; ++mt)
#pragma unroll
            for (int r = 0; r < 8; ++r) {
                int mrow = m0 + mt * 16 + r + hf * 8;
                int bb = mrow / NSEQ;
                int nn = mrow - bb * NSEQ;
                bf16_t* base = dst + ((size_t)(bb * NH + head) * NPAD + nn) * HDIM;
#pragma unroll
                for (int t = 0; t < 4; ++t) base[t * 16 + l16] = (bf16_t)acc[mt][t][r];
            }
    }
}

// ---------------------------------------------------------------- flash attention
// grid (17, 256), block 128 (4 waves). One wave = one 16-row q tile of one (b,h).
__global__ __launch_bounds__(128) void attn_kernel(
    const bf16_t* __restrict__ Q, const bf16_t* __restrict__ K,
    const bf16_t* __restrict__ VT, bf16_t* __restrict__ ctx)
{
    __shared__ alignas(16) bf16_t pbuf[4][16][32];
    const int lane = threadIdx.x & 31;
    const int wv   = threadIdx.x >> 5;
    const int l16  = lane & 15;
    const int hf   = lane >> 4;
    const int qt   = blockIdx.x * 4 + wv;
    if (qt >= 65) return;                       // no block barriers used below
    const int bh = blockIdx.y;

    const bf16_t* Qb = Q  + (size_t)bh * NPAD * HDIM;
    const bf16_t* Kb = K  + (size_t)bh * NPAD * HDIM;
    const bf16_t* Vb = VT + (size_t)bh * HDIM * NPAD;

    const int q0 = qt * 16;
    const bf16_t* qrow = Qb + (size_t)(q0 + l16) * HDIM;
    v16bf a_lo = ldA(qrow, 0, hf);    // d = 0..31
    v16bf a_hi = ldA(qrow, 32, hf);   // d = 32..63

    float m[8], l[8];
#pragma unroll
    for (int r = 0; r < 8; ++r) { m[r] = -INFINITY; l[r] = 0.f; }
    v8f o[4] = {};

    for (int k0 = 0; k0 < NPAD; k0 += 32) {
        const bf16_t* kr0 = Kb + (size_t)(k0 + l16) * HDIM;
        const bf16_t* kr1 = Kb + (size_t)(k0 + 16 + l16) * HDIM;
        v8f s0 = {}, s1 = {};
        s0 = wmma_bf16(a_lo, ldB(kr0 + hf * 16),      s0);
        s0 = wmma_bf16(a_hi, ldB(kr0 + 32 + hf * 16), s0);
        s1 = wmma_bf16(a_lo, ldB(kr1 + hf * 16),      s1);
        s1 = wmma_bf16(a_hi, ldB(kr1 + 32 + hf * 16), s1);

        const bool ok0 = (k0 + l16)      < NSEQ;
        const bool ok1 = (k0 + 16 + l16) < NSEQ;
        float alpha[8];
#pragma unroll
        for (int r = 0; r < 8; ++r) {
            float v0 = ok0 ? s0[r] * 0.125f : -INFINITY;
            float v1 = ok1 ? s1[r] * 0.125f : -INFINITY;
            float rm = fmaxf(v0, v1);
#pragma unroll
            for (int off = 1; off < 16; off <<= 1)
                rm = fmaxf(rm, __shfl_xor(rm, off, 32));
            float mn = fmaxf(m[r], rm);
            float al = __expf(m[r] - mn);
            float p0 = __expf(v0 - mn);
            float p1 = __expf(v1 - mn);
            float rs = p0 + p1;
#pragma unroll
            for (int off = 1; off < 16; off <<= 1)
                rs += __shfl_xor(rs, off, 32);
            l[r] = l[r] * al + rs;
            m[r] = mn;
            alpha[r] = al;
            s0[r] = p0; s1[r] = p1;
        }
#pragma unroll
        for (int t = 0; t < 4; ++t)
#pragma unroll
            for (int r = 0; r < 8; ++r) o[t][r] *= alpha[r];

        // C-layout -> A-layout via LDS (wave-private tile; DS ops are in-order per wave)
#pragma unroll
        for (int r = 0; r < 8; ++r) {
            int rr = r + hf * 8;
            pbuf[wv][rr][l16]      = (bf16_t)s0[r];
            pbuf[wv][rr][16 + l16] = (bf16_t)s1[r];
        }
        const v8bf* prow = reinterpret_cast<const v8bf*>(&pbuf[wv][l16][0]);
        v16bf ap = mk16(prow[2 * hf], prow[2 * hf + 1]);

#pragma unroll
        for (int t = 0; t < 4; ++t) {
            const bf16_t* vrow = Vb + (size_t)(t * 16 + l16) * NPAD + k0 + hf * 16;
            o[t] = wmma_bf16(ap, ldB(vrow), o[t]);
        }
    }

#pragma unroll
    for (int r = 0; r < 8; ++r) {
        float inv = 1.0f / l[r];
#pragma unroll
        for (int t = 0; t < 4; ++t) o[t][r] *= inv;
    }
    const int b = bh >> 4, h = bh & 15;
#pragma unroll
    for (int r = 0; r < 8; ++r) {
        int nrow = q0 + r + hf * 8;
        if (nrow < NSEQ) {
            bf16_t* dst = ctx + ((size_t)(b * NSEQ + nrow)) * DMODEL + h * 64;
#pragma unroll
            for (int t = 0; t < 4; ++t) dst[t * 16 + l16] = (bf16_t)o[t][r];
        }
    }
}

// ---------------------------------------------------------------- output projection
// grid (343, 2), block 256 (8 waves). Wave = 48(M) x 64(N) tile.
__global__ __launch_bounds__(256) void oproj_kernel(
    const bf16_t* __restrict__ Cbf, const bf16_t* __restrict__ Wo,
    const float* __restrict__ bo, float* __restrict__ out)
{
    const int lane = threadIdx.x & 31;
    const int wv   = threadIdx.x >> 5;
    const int l16  = lane & 15;
    const int hf   = lane >> 4;
    const int n0   = (blockIdx.y * 8 + wv) * 64;
    const int m0   = blockIdx.x * 48;

    v8f acc[3][4] = {};
    for (int k0 = 0; k0 < DMODEL; k0 += 32) {
        v16bf a[3];
#pragma unroll
        for (int mt = 0; mt < 3; ++mt)
            a[mt] = ldA(Cbf + (size_t)(m0 + mt * 16 + l16) * DMODEL, k0, hf);
#pragma unroll
        for (int t = 0; t < 4; ++t) {
            const bf16_t* wrow = Wo + (size_t)(n0 + t * 16 + l16) * DMODEL + k0 + hf * 16;
            v16bf b = ldB(wrow);
#pragma unroll
            for (int mt = 0; mt < 3; ++mt)
                acc[mt][t] = wmma_bf16(a[mt], b, acc[mt][t]);
        }
    }
#pragma unroll
    for (int t = 0; t < 4; ++t) {
        float bb = bo[n0 + t * 16 + l16];
        float* col = out + n0 + t * 16 + l16;
#pragma unroll
        for (int mt = 0; mt < 3; ++mt)
#pragma unroll
            for (int r = 0; r < 8; ++r) {
                int mrow = m0 + mt * 16 + r + hf * 8;
                col[(size_t)mrow * DMODEL] = acc[mt][t][r] + bb;
            }
    }
}

// ---------------------------------------------------------------- launch
extern "C" void kernel_launch(void* const* d_in, const int* in_sizes, int n_in,
                              void* d_out, int out_size, void* d_ws, size_t ws_size,
                              hipStream_t stream) {
    (void)in_sizes; (void)n_in; (void)out_size; (void)ws_size;
    const float* hs   = (const float*)d_in[0];
    const float* sinp = (const float*)d_in[1];
    const float* cosp = (const float*)d_in[2];
    const float* Wq   = (const float*)d_in[3];
    const float* bq   = (const float*)d_in[4];
    const float* Wk   = (const float*)d_in[5];
    const float* bk   = (const float*)d_in[6];
    const float* Wv   = (const float*)d_in[7];
    const float* bv   = (const float*)d_in[8];
    const float* Wo   = (const float*)d_in[9];
    const float* bo   = (const float*)d_in[10];

    uint8_t* ws = (uint8_t*)d_ws;
    const size_t SZ_X = (size_t)MTOT * DMODEL * sizeof(bf16_t);          // 33.7 MB
    const size_t SZ_W = (size_t)4 * DMODEL * DMODEL * sizeof(bf16_t);    //  8.4 MB
    const size_t SZ_P = (size_t)NB * NH * NPAD * HDIM * sizeof(bf16_t);  // 34.6 MB
    bf16_t* Xbf = (bf16_t*)(ws);
    bf16_t* Wbf = (bf16_t*)(ws + SZ_X);
    bf16_t* Qw  = (bf16_t*)(ws + SZ_X + SZ_W);
    bf16_t* Kw  = (bf16_t*)(ws + SZ_X + SZ_W + SZ_P);
    bf16_t* VTw = (bf16_t*)(ws + SZ_X + SZ_W + 2 * SZ_P);
    bf16_t* Cw  = (bf16_t*)(ws + SZ_X + SZ_W + 3 * SZ_P);

    // 1) convert fp32 -> bf16 (X and the 4 weight matrices)
    {
        int n4 = MTOT * DMODEL / 4;
        cvt_bf16_kernel<<<(n4 + 255) / 256, 256, 0, stream>>>(hs, Xbf, n4);
        int w4 = DMODEL * DMODEL / 4;
        cvt_bf16_kernel<<<(w4 + 255) / 256, 256, 0, stream>>>(Wq, Wbf + 0 * (size_t)DMODEL * DMODEL, w4);
        cvt_bf16_kernel<<<(w4 + 255) / 256, 256, 0, stream>>>(Wk, Wbf + 1 * (size_t)DMODEL * DMODEL, w4);
        cvt_bf16_kernel<<<(w4 + 255) / 256, 256, 0, stream>>>(Wv, Wbf + 2 * (size_t)DMODEL * DMODEL, w4);
        cvt_bf16_kernel<<<(w4 + 255) / 256, 256, 0, stream>>>(Wo, Wbf + 3 * (size_t)DMODEL * DMODEL, w4);
    }
    // 2) fused QKV projection + bias + RoPE, head-split stores (V transposed)
    qkv_proj_kernel<<<dim3(MTOT / 48, 6), 256, 0, stream>>>(
        Xbf, Wbf, bq, bk, bv, sinp, cosp, Qw, Kw, VTw);
    // 3) flash attention per (b,h)
    attn_kernel<<<dim3(17, NB * NH), 128, 0, stream>>>(Qw, Kw, VTw, Cw);
    // 4) output projection + bias -> f32 output
    oproj_kernel<<<dim3(MTOT / 48, 2), 256, 0, stream>>>(
        Cw, Wbf + 3 * (size_t)DMODEL * DMODEL, bo, (float*)d_out);
}
